// cnv_48962627175297
// MI455X (gfx1250) — 8-bit matrix operand order repaired and checked by reading, compiled, not run on hardware
//
#include <hip/hip_runtime.h>
#include <cstdint>
#include <cstddef>


typedef __attribute__((ext_vector_type(8)))  int   v8i;
typedef __attribute__((ext_vector_type(16))) int   v16i;
typedef __attribute__((ext_vector_type(8)))  float v8f;

#define FP8_P1 ((uint8_t)0x38)
#define FP8_M1 ((uint8_t)0xB8)
#define EPS_BN 1e-5f

template<int CH> struct AFrag;
template<> struct AFrag<128> { v16i v; };
template<> struct AFrag<64>  { v8i  v; };

template<int CH>
__device__ __forceinline__ void load_afrag(const uint8_t* p, AFrag<CH>& f) {
  if constexpr (CH == 128) {
    union { v16i v; int2 d[8]; } u;
#pragma unroll
    for (int j = 0; j < 8; ++j)
      u.d[j] = *(const int2*)(p + (((j >> 1) << 5) + ((j & 1) << 3)));
    f.v = u.v;
  } else {
    union { v8i v; int2 d[4]; } u;
#pragma unroll
    for (int j = 0; j < 4; ++j)
      u.d[j] = *(const int2*)(p + (((j >> 1) << 5) + ((j & 1) << 3)));
    f.v = u.v;
  }
}

template<int CH>
__device__ __forceinline__ void load_bfrag(const uint8_t* p, AFrag<CH>& f) {
  if constexpr (CH == 128) {
    union { v16i v; int4 q[4]; } u;
#pragma unroll
    for (int g = 0; g < 4; ++g)
      u.q[g] = *(const int4*)(p + (g << 5));
    f.v = u.v;
  } else {
    union { v8i v; int4 q[2]; } u;
#pragma unroll
    for (int g = 0; g < 2; ++g)
      u.q[g] = *(const int4*)(p + (g << 5));
    f.v = u.v;
  }
}

template<int CH>
__device__ __forceinline__ v8f wmma_fp8(const AFrag<CH>& a, const AFrag<CH>& b, v8f c) {
  if constexpr (CH == 128)
    return __builtin_amdgcn_wmma_f32_16x16x128_fp8_fp8(a.v, b.v, (short)0, c, false, false);
  else
    return __builtin_amdgcn_wmma_f32_16x16x64_fp8_fp8(a.v, b.v, (short)0, c, false, false);
}


template<int CI, int CO>
__global__ void __launch_bounds__(256)
prep_conv_w(const float* __restrict__ w, uint8_t* __restrict__ bw) {
  const int K = CI * 9;
  int idx = blockIdx.x * 256 + threadIdx.x;
  if (idx >= CO * K) return;
  int co = idx / K;
  int r  = idx - co * K;
  int e  = r / CI;
  int ci = r - e * CI;
  int kh = e / 3, kw = e - kh * 3;
  float val = w[((co * CI + ci) * 3 + kh) * 3 + kw];
  bw[idx] = (val >= 0.f) ? FP8_P1 : FP8_M1;
}

__global__ void __launch_bounds__(256)
prep_fc0_w(const float* __restrict__ w, uint8_t* __restrict__ bw) {
  int idx = blockIdx.x * 256 + threadIdx.x;
  if (idx >= 512 * 20736) return;
  int co = idx / 20736;
  int kp = idx - co * 20736;
  int hw = kp >> 8;
  int c  = kp & 255;
  int k  = c * 81 + hw;
  bw[idx] = (w[co * 20736 + k] >= 0.f) ? FP8_P1 : FP8_M1;
}

template<int K, int CO, int COPAD>
__global__ void __launch_bounds__(256)
prep_fc_w(const float* __restrict__ w, uint8_t* __restrict__ bw) {
  int idx = blockIdx.x * 256 + threadIdx.x;
  if (idx >= COPAD * K) return;
  int co = idx / K;
  int k  = idx - co * K;
  bw[idx] = (co < CO) ? ((w[co * K + k] >= 0.f) ? FP8_P1 : FP8_M1) : (uint8_t)0x00;
}

__global__ void __launch_bounds__(64)
conv0_kernel(const float* __restrict__ x, const float* __restrict__ w,
             const float* __restrict__ gg, const float* __restrict__ bt,
             const float* __restrict__ mn, const float* __restrict__ vr,
             uint8_t* __restrict__ out) {
  const int pix = blockIdx.x;
  const int wo = pix % 62;
  const int t  = pix / 62;
  const int ho = t % 62;
  const int bb = t / 62;
  const int co = threadIdx.x;
  __shared__ float xs[27];
  if (threadIdx.x < 27) {
    int ci = threadIdx.x / 9;
    int r  = threadIdx.x - ci * 9;
    int kh = r / 3, kw = r - kh * 3;
    xs[threadIdx.x] = x[(((size_t)bb * 3 + ci) * 64 + (ho + kh)) * 64 + (wo + kw)];
  }
  __syncthreads();
  float acc = 0.f;
#pragma unroll
  for (int r = 0; r < 27; ++r) {
    float wv = w[co * 27 + r];
    acc += xs[r] * ((wv >= 0.f) ? 1.f : -1.f);
  }
  float sc = rsqrtf(vr[co] + EPS_BN) * gg[co];
  float val = (acc - mn[co]) * sc + bt[co];
  out[(size_t)pix * 64 + co] = (val >= 0.f) ? FP8_P1 : FP8_M1;
}

template<int CI, int CO, int H, int W, int HO, int WO, int NT>
__global__ void __launch_bounds__(256)
bconv_gemm(const uint8_t* __restrict__ act, const uint8_t* __restrict__ bw,
           const float* __restrict__ gg, const float* __restrict__ bt,
           const float* __restrict__ mn, const float* __restrict__ vr,
           uint8_t* __restrict__ out) {
  constexpr int K     = CI * 9;
  constexpr int CH    = (CI == 64) ? 64 : 128;
  constexpr int PERKH = (3 * CI) / CH;
  constexpr int NCH   = 3 * PERKH;

  const int lane = threadIdx.x & 31;
  const int wv   = threadIdx.x >> 5;
  const int hi   = lane >> 4;
  const int lm   = lane & 15;
  const uint32_t m0 = (blockIdx.x * 8 + wv) * 16;
  const uint32_t n0 = blockIdx.y * (16 * NT);

  const uint32_t row = m0 + lm;
  const uint32_t b_  = row / (HO * WO);
  const uint32_t rr  = row - b_ * (HO * WO);
  const uint32_t ho  = rr / WO;
  const uint32_t wo  = rr - ho * WO;
  const uint8_t* pA = act + (((b_ * H + ho) * W + wo) * CI + (hi << 4));
  const uint8_t* segA[3] = { pA, pA + W * CI, pA + 2 * W * CI };
  const uint8_t* pB[NT];
#pragma unroll
  for (int t = 0; t < NT; ++t)
    pB[t] = bw + ((n0 + t * 16u + lm) * (uint32_t)K + (hi << 4));

  const v8f vzero = {};
  v8f acc[NT];
#pragma unroll
  for (int t = 0; t < NT; ++t) acc[t] = vzero;

  AFrag<CH> a, b[NT];
  load_afrag<CH>(segA[0], a);
#pragma unroll
  for (int t = 0; t < NT; ++t) load_bfrag<CH>(pB[t], b[t]);

#pragma unroll
  for (int c = 0; c < NCH; ++c) {
    AFrag<CH> an, bn[NT];
    if (c + 1 < NCH) {
      load_afrag<CH>(segA[(c + 1) / PERKH] + ((c + 1) % PERKH) * CH, an);
#pragma unroll
      for (int t = 0; t < NT; ++t)
        load_bfrag<CH>(pB[t] + (c + 1) * CH, bn[t]);
    }
#pragma unroll
    for (int t = 0; t < NT; ++t)
      acc[t] = wmma_fp8<CH>(a, b[t], acc[t]);
    a = an;
#pragma unroll
    for (int t = 0; t < NT; ++t) b[t] = bn[t];
  }

#pragma unroll
  for (int t = 0; t < NT; ++t) {
    const uint32_t n = n0 + t * 16 + lm;
    const float sc = rsqrtf(vr[n] + EPS_BN) * gg[n];
    const float sh = bt[n] - mn[n] * sc;
    union { v8f v; float f[8]; } cc; cc.v = acc[t];
#pragma unroll
    for (int r = 0; r < 8; ++r) {
      const uint32_t mrow = m0 + r + 8 * hi;
      const float val = cc.f[r] * sc + sh;
      out[mrow * (uint32_t)CO + n] = (val >= 0.f) ? FP8_P1 : FP8_M1;
    }
  }
}

template<int K, int COPAD, int COV, int NT, bool SIGN>
__global__ void __launch_bounds__(256)
bfc_gemm(const uint8_t* __restrict__ act, const uint8_t* __restrict__ bw,
         const float* __restrict__ bias,
         const float* __restrict__ gg, const float* __restrict__ bt,
         const float* __restrict__ mn, const float* __restrict__ vr,
         uint8_t* __restrict__ out8, float* __restrict__ outf) {
  static_assert(K % 128 == 0, "K must be a multiple of 128");
  constexpr int NCH = K / 128;
  const int lane = threadIdx.x & 31;
  const int wv   = threadIdx.x >> 5;
  const int hi   = lane >> 4;
  const int lm   = lane & 15;
  const uint32_t m0 = (blockIdx.x * 8 + wv) * 16;
  const uint32_t n0 = blockIdx.y * (16 * NT);

  const uint8_t* pa = act + ((m0 + lm) * (uint32_t)K + (hi << 4));
  const uint8_t* pB[NT];
#pragma unroll
  for (int t = 0; t < NT; ++t)
    pB[t] = bw + ((n0 + t * 16u + lm) * (uint32_t)K + (hi << 4));

  const v8f vzero = {};
  v8f acc[NT];
#pragma unroll
  for (int t = 0; t < NT; ++t) acc[t] = vzero;

  AFrag<128> a, b[NT];
  load_afrag<128>(pa, a);
#pragma unroll
  for (int t = 0; t < NT; ++t) load_bfrag<128>(pB[t], b[t]);

  for (int c = 1; c < NCH; ++c) {
    __builtin_prefetch(pa + c * 128 + 1024, 0, 1);
    AFrag<128> an, bn[NT];
    load_afrag<128>(pa + c * 128, an);
#pragma unroll
    for (int t = 0; t < NT; ++t) load_bfrag<128>(pB[t] + c * 128, bn[t]);
#pragma unroll
    for (int t = 0; t < NT; ++t) acc[t] = wmma_fp8<128>(a, b[t], acc[t]);
    a = an;
#pragma unroll
    for (int t = 0; t < NT; ++t) b[t] = bn[t];
  }
#pragma unroll
  for (int t = 0; t < NT; ++t) acc[t] = wmma_fp8<128>(a, b[t], acc[t]);

#pragma unroll
  for (int t = 0; t < NT; ++t) {
    const uint32_t n = n0 + t * 16 + lm;
    if (n < (uint32_t)COV) {
      const float sc = rsqrtf(vr[n] + EPS_BN) * gg[n];
      const float sh = bt[n] - mn[n] * sc;
      const float bs = bias[n];
      union { v8f v; float f[8]; } cc; cc.v = acc[t];
#pragma unroll
      for (int r = 0; r < 8; ++r) {
        const uint32_t mrow = m0 + r + 8 * hi;
        const float val = (cc.f[r] + bs) * sc + sh;
        if (SIGN)
          out8[mrow * (uint32_t)COPAD + n] = (val >= 0.f) ? FP8_P1 : FP8_M1;
        else
          outf[mrow * (uint32_t)COPAD + n] = val;
      }
    }
  }
}

__global__ void __launch_bounds__(256)
pool_kernel(const uint8_t* __restrict__ in, uint8_t* __restrict__ out,
            int C, int HI, int WI, int n_total) {
  int idx = blockIdx.x * 256 + threadIdx.x;
  if (idx >= n_total) return;
  int c = idx % C;
  int t = idx / C;
  int WO = WI >> 1, HO = HI >> 1;
  int wo = t % WO; t /= WO;
  int ho = t % HO;
  int bb = t / HO;
  const uint8_t* p = in + (((size_t)bb * HI + 2 * ho) * WI + 2 * wo) * C + c;
  int8_t a0 = (int8_t)p[0];
  int8_t a1 = (int8_t)p[C];
  int8_t a2 = (int8_t)p[(size_t)WI * C];
  int8_t a3 = (int8_t)p[(size_t)WI * C + C];
  int8_t mx = a0 > a1 ? a0 : a1;
  int8_t my = a2 > a3 ? a2 : a3;
  out[idx] = (uint8_t)(mx > my ? mx : my);
}

__global__ void __launch_bounds__(256)
lsm_kernel(const float* __restrict__ logits, float* __restrict__ out) {
  int bb = blockIdx.x * blockDim.x + threadIdx.x;
  if (bb >= 256) return;
  const float* r = logits + (size_t)bb * 16;
  float mx = r[0];
#pragma unroll
  for (int j = 1; j < 10; ++j) mx = fmaxf(mx, r[j]);
  float s = 0.f;
#pragma unroll
  for (int j = 0; j < 10; ++j) s += expf(r[j] - mx);
  float lse = mx + logf(s);
#pragma unroll
  for (int j = 0; j < 10; ++j) out[bb * 10 + j] = r[j] - lse;
}


extern "C" void kernel_launch(void* const* d_in, const int* in_sizes, int n_in,
                              void* d_out, int out_size, void* d_ws, size_t ws_size,
                              hipStream_t stream) {
  (void)in_sizes; (void)n_in; (void)out_size; (void)ws_size;
  const float* x = (const float*)d_in[0];
  const float *cw[6], *fw[3], *fb[3], *G[9], *Bt[9], *Mn[9], *Vr[9];
  for (int i = 0; i < 6; ++i) {
    cw[i] = (const float*)d_in[1 + i * 5 + 0];
    G[i]  = (const float*)d_in[1 + i * 5 + 1];
    Bt[i] = (const float*)d_in[1 + i * 5 + 2];
    Mn[i] = (const float*)d_in[1 + i * 5 + 3];
    Vr[i] = (const float*)d_in[1 + i * 5 + 4];
  }
  for (int j = 0; j < 3; ++j) {
    int base = 31 + j * 6;
    fw[j]    = (const float*)d_in[base + 0];
    fb[j]    = (const float*)d_in[base + 1];
    G[6 + j] = (const float*)d_in[base + 2];
    Bt[6 + j]= (const float*)d_in[base + 3];
    Mn[6 + j]= (const float*)d_in[base + 4];
    Vr[6 + j]= (const float*)d_in[base + 5];
  }

  uint8_t* ws   = (uint8_t*)d_ws;
  uint8_t* bufA = ws;
  uint8_t* bufB = ws + ((size_t)64 << 20);
  uint8_t* wb   = ws + ((size_t)128 << 20);
  uint8_t* w1  = wb + 0;
  uint8_t* w2  = wb + 40960;
  uint8_t* w3  = wb + 114688;
  uint8_t* w4  = wb + 262144;
  uint8_t* w5  = wb + 557056;
  uint8_t* wf0 = wb + 1146880;
  uint8_t* wf1 = wb + 11763712;
  uint8_t* wf2 = wb + 12025856;
  float* logits = (float*)(wb + 12034048);

  prep_conv_w<64, 64 ><<<144,  256, 0, stream>>>(cw[1], w1);
  prep_conv_w<64, 128><<<288,  256, 0, stream>>>(cw[2], w2);
  prep_conv_w<128,128><<<576,  256, 0, stream>>>(cw[3], w3);
  prep_conv_w<128,256><<<1152, 256, 0, stream>>>(cw[4], w4);
  prep_conv_w<256,256><<<2304, 256, 0, stream>>>(cw[5], w5);
  prep_fc0_w<<<41472, 256, 0, stream>>>(fw[0], wf0);
  prep_fc_w<512, 512, 512><<<1024, 256, 0, stream>>>(fw[1], wf1);
  prep_fc_w<512, 10,  16 ><<<32,   256, 0, stream>>>(fw[2], wf2);

  conv0_kernel<<<256 * 62 * 62, 64, 0, stream>>>(
      x, cw[0], G[0], Bt[0], Mn[0], Vr[0], bufA);

  bconv_gemm<64, 64, 62, 62, 60, 60, 2><<<dim3(7200, 2), 256, 0, stream>>>(
      bufA, w1, G[1], Bt[1], Mn[1], Vr[1], bufB);
  pool_kernel<<<57600, 256, 0, stream>>>(bufB, bufA, 64, 60, 60, 14745600);

  bconv_gemm<64, 128, 30, 30, 28, 28, 2><<<dim3(1568, 4), 256, 0, stream>>>(
      bufA, w2, G[2], Bt[2], Mn[2], Vr[2], bufB);
  bconv_gemm<128, 128, 28, 28, 26, 26, 2><<<dim3(1352, 4), 256, 0, stream>>>(
      bufB, w3, G[3], Bt[3], Mn[3], Vr[3], bufA);
  pool_kernel<<<21632, 256, 0, stream>>>(bufA, bufB, 128, 26, 26, 5537792);

  bconv_gemm<128, 256, 13, 13, 11, 11, 2><<<dim3(242, 8), 256, 0, stream>>>(
      bufB, w4, G[4], Bt[4], Mn[4], Vr[4], bufA);
  bconv_gemm<256, 256, 11, 11, 9, 9, 2><<<dim3(162, 8), 256, 0, stream>>>(
      bufA, w5, G[5], Bt[5], Mn[5], Vr[5], bufB);

  bfc_gemm<20736, 512, 512, 2, true><<<dim3(2, 16), 256, 0, stream>>>(
      bufB, wf0, fb[0], G[6], Bt[6], Mn[6], Vr[6], bufA, nullptr);
  bfc_gemm<512, 512, 512, 2, true><<<dim3(2, 16), 256, 0, stream>>>(
      bufA, wf1, fb[1], G[7], Bt[7], Mn[7], Vr[7], bufB, nullptr);
  bfc_gemm<512, 16, 10, 1, false><<<dim3(2, 1), 256, 0, stream>>>(
      bufB, wf2, fb[2], G[8], Bt[8], Mn[8], Vr[8], nullptr, logits);

  lsm_kernel<<<1, 256, 0, stream>>>(logits, (float*)d_out);
}
